// _DiscreteLSTMGenerator_8641474200356
// MI455X (gfx1250) — matrix operand order repaired and checked by reading, not compiled, not run on hardware
//
#include <hip/hip_runtime.h>
#include <hip/hip_bf16.h>
#include <cstdint>
#include <cstddef>

#define VOCAB   2000
#define SEQ_LEN 128
#define EMB     512
#define HID     1024
#define BATCH   256
#define GATES_N (4 * HID)   // 4096

typedef __bf16 bf16_t;
typedef __attribute__((ext_vector_type(16))) __bf16 v16bf;
typedef __attribute__((ext_vector_type(8)))  float  v8f;

// ---------------------------------------------------------------------------
// WMMA fragment loaders (wave32, contiguous k = 16*(lane/16) + elem in both A and B, CDNA5 ISA §7.12.2)
// ---------------------------------------------------------------------------

// A-matrix 16x32 bf16, row-major activation [M, lda].
// lanes 0-15: row M=lane,    elems 0..15 -> K=k0..k0+15
// lanes 16-31: row M=lane-16, elems 0..15 -> K=k0+16..k0+31
__device__ __forceinline__ v16bf load_a_frag(const bf16_t* __restrict__ A,
                                             int lda, int m0, int k0) {
  const int lane = threadIdx.x & 31;
  const int hs   = lane >> 4;
  const int row  = m0 + (lane & 15);
  const bf16_t* p1 = A + (size_t)row * lda + k0 + 16 * hs;
  const bf16_t* p2 = p1 + 8;
  v16bf a;
#pragma unroll
  for (int i = 0; i < 8; ++i) { a[i] = p1[i]; a[8 + i] = p2[i]; }
  return a;
}

// B-matrix 32x16 = (W tile)^T where W is row-major [N, ldw].
// lane n (0-15):  column n0+n, elems = K k0..k0+15   (row n0+n of W, contiguous)
// lane n+16:      column n0+n, elems = K k0+16..k0+31
__device__ __forceinline__ v16bf load_b_frag(const bf16_t* __restrict__ W,
                                             int ldw, int n0, int k0) {
  const int lane = threadIdx.x & 31;
  const int col  = n0 + (lane & 15);
  const bf16_t* p = W + (size_t)col * ldw + k0 + 16 * (lane >> 4);
  __builtin_prefetch(p + 128, 0, 0);   // global_prefetch_b8: pull L2 weights ahead
  v16bf b;
#pragma unroll
  for (int i = 0; i < 16; ++i) b[i] = p[i];
  return b;
}

__device__ __forceinline__ v8f wmma_bf16(v16bf a, v16bf b, v8f c) {
  return __builtin_amdgcn_wmma_f32_16x16x32_bf16(false, a, false, b,
                                                 (short)0, c, false, false);
}

// ---------------------------------------------------------------------------
// Kernel 1: gates = x_bf @ W_ih^T + h_bf @ W_hh^T + (b_ih + b_hh)   [256,4096]
// Register blocking: 2 M-tiles x 4 N-tiles per wave (8 WMMA per K-step from
// 2 A + 4 B fragment loads). 8 M-groups x 64 N-groups = 512 waves = 64 blocks.
// ---------------------------------------------------------------------------
__global__ __launch_bounds__(256) void gates_gemm_kernel(
    const bf16_t* __restrict__ x_bf, const bf16_t* __restrict__ h_bf,
    const bf16_t* __restrict__ Wih,  const bf16_t* __restrict__ Whh,
    const float*  __restrict__ b_ih, const float*  __restrict__ b_hh,
    float* __restrict__ gates) {
  const int wave = threadIdx.x >> 5;
  const int g    = blockIdx.x * 8 + wave;   // 0..511
  const int mg   = g >> 6;                  // 0..7   -> m0 = mg*32
  const int ngr  = g & 63;                  // 0..63  -> n0 = ngr*64
  const int m0 = mg * 32;
  const int n0 = ngr * 64;
  const int lane  = threadIdx.x & 31;
  const int nlane = lane & 15;

  v8f acc[2][4];
#pragma unroll
  for (int j = 0; j < 4; ++j) {
    const int n = n0 + j * 16 + nlane;
    const float bias = b_ih[n] + b_hh[n];
#pragma unroll
    for (int i = 0; i < 2; ++i)
#pragma unroll
      for (int r = 0; r < 8; ++r) acc[i][j][r] = bias;
  }

  for (int k0 = 0; k0 < EMB; k0 += 32) {
    v16bf a0 = load_a_frag(x_bf, EMB, m0,      k0);
    v16bf a1 = load_a_frag(x_bf, EMB, m0 + 16, k0);
#pragma unroll
    for (int j = 0; j < 4; ++j) {
      v16bf b = load_b_frag(Wih, EMB, n0 + j * 16, k0);
      acc[0][j] = wmma_bf16(a0, b, acc[0][j]);
      acc[1][j] = wmma_bf16(a1, b, acc[1][j]);
    }
  }
  for (int k0 = 0; k0 < HID; k0 += 32) {
    v16bf a0 = load_a_frag(h_bf, HID, m0,      k0);
    v16bf a1 = load_a_frag(h_bf, HID, m0 + 16, k0);
#pragma unroll
    for (int j = 0; j < 4; ++j) {
      v16bf b = load_b_frag(Whh, HID, n0 + j * 16, k0);
      acc[0][j] = wmma_bf16(a0, b, acc[0][j]);
      acc[1][j] = wmma_bf16(a1, b, acc[1][j]);
    }
  }

  const int roff = 8 * (lane >> 4);
#pragma unroll
  for (int i = 0; i < 2; ++i) {
    const int rbase = m0 + i * 16 + roff;
#pragma unroll
    for (int j = 0; j < 4; ++j) {
      const int n = n0 + j * 16 + nlane;
#pragma unroll
      for (int r = 0; r < 8; ++r)
        gates[(size_t)(rbase + r) * GATES_N + n] = acc[i][j][r];
    }
  }
}

// ---------------------------------------------------------------------------
// Kernel 2: LSTM cell elementwise (PyTorch gate order i,f,g,o)
// ---------------------------------------------------------------------------
__global__ __launch_bounds__(256) void lstm_cell_kernel(
    const float* __restrict__ gates, float* __restrict__ c,
    bf16_t* __restrict__ h_bf) {
  const int idx = blockIdx.x * 256 + threadIdx.x;   // < BATCH*HID
  const int b = idx >> 10;
  const int j = idx & (HID - 1);
  const float* g = gates + (size_t)b * GATES_N;
  const float gi = g[j];
  const float gf = g[HID + j];
  const float gg = g[2 * HID + j];
  const float go = g[3 * HID + j];
  const float si = 1.f / (1.f + __expf(-gi));
  const float sf = 1.f / (1.f + __expf(-gf));
  const float so = 1.f / (1.f + __expf(-go));
  const float cc = sf * c[idx] + si * tanhf(gg);
  c[idx] = cc;
  h_bf[idx] = (bf16_t)(so * tanhf(cc));
}

// ---------------------------------------------------------------------------
// Kernel 3: logits = h_bf @ W_out^T + b_out + gumbel[s]   [256,2000] (fp32)
// Register blocking: 2 M-tiles x 5 N-tiles per wave.
// 8 M-groups x 25 N-groups = 200 waves = 25 blocks (exact).
// ---------------------------------------------------------------------------
__global__ __launch_bounds__(256) void logits_gemm_kernel(
    const bf16_t* __restrict__ h_bf, const bf16_t* __restrict__ Wout,
    const float* __restrict__ b_out, const float* __restrict__ gumbel_s,
    float* __restrict__ logits) {
  const int wave = threadIdx.x >> 5;
  const int g    = blockIdx.x * 8 + wave;   // 0..199
  const int mg   = g / 25;                  // 0..7   -> m0 = mg*32
  const int ngr  = g % 25;                  // 0..24  -> n0 = ngr*80
  const int m0 = mg * 32;
  const int n0 = ngr * 80;
  const int lane  = threadIdx.x & 31;
  const int nlane = lane & 15;

  v8f acc[2][5];
#pragma unroll
  for (int j = 0; j < 5; ++j) {
    const float bias = b_out[n0 + j * 16 + nlane];
#pragma unroll
    for (int i = 0; i < 2; ++i)
#pragma unroll
      for (int r = 0; r < 8; ++r) acc[i][j][r] = bias;
  }

  for (int k0 = 0; k0 < HID; k0 += 32) {
    v16bf a0 = load_a_frag(h_bf, HID, m0,      k0);
    v16bf a1 = load_a_frag(h_bf, HID, m0 + 16, k0);
#pragma unroll
    for (int j = 0; j < 5; ++j) {
      v16bf b = load_b_frag(Wout, HID, n0 + j * 16, k0);
      acc[0][j] = wmma_bf16(a0, b, acc[0][j]);
      acc[1][j] = wmma_bf16(a1, b, acc[1][j]);
    }
  }

  const int roff = 8 * (lane >> 4);
#pragma unroll
  for (int i = 0; i < 2; ++i) {
    const int rbase = m0 + i * 16 + roff;
#pragma unroll
    for (int j = 0; j < 5; ++j) {
      const int n = n0 + j * 16 + nlane;
#pragma unroll
      for (int r = 0; r < 8; ++r) {
        const int m = rbase + r;
        logits[(size_t)m * VOCAB + n] = acc[i][j][r] + gumbel_s[(size_t)m * VOCAB + n];
      }
    }
  }
}

// ---------------------------------------------------------------------------
// Kernel 4: per-batch-row argmax -> one-hot row into d_out, gather emb row
// as next x (bf16). Tie-break = lowest index (matches jnp.argmax).
// ---------------------------------------------------------------------------
__global__ __launch_bounds__(256) void argmax_onehot_kernel(
    const float* __restrict__ logits, const float* __restrict__ emb_W,
    float* __restrict__ out, bf16_t* __restrict__ x_bf, int s) {
  const int b = blockIdx.x;
  const int tid = threadIdx.x;
  __shared__ float sval[256];
  __shared__ int   sidx[256];

  float best = -3.402823466e38f;
  int   bi   = 0;
  const float* row = logits + (size_t)b * VOCAB;
  for (int v = tid; v < VOCAB; v += 256) {
    const float val = row[v];
    if (val > best) { best = val; bi = v; }   // strict > keeps first index
  }
  sval[tid] = best; sidx[tid] = bi;
  __syncthreads();
#pragma unroll
  for (int stride = 128; stride > 0; stride >>= 1) {
    if (tid < stride) {
      const float ov = sval[tid + stride];
      const int   oi = sidx[tid + stride];
      if (ov > sval[tid] || (ov == sval[tid] && oi < sidx[tid])) {
        sval[tid] = ov; sidx[tid] = oi;
      }
    }
    __syncthreads();
  }
  const int amax = sidx[0];

  float* orow = out + (size_t)b * SEQ_LEN * VOCAB + (size_t)s * VOCAB;
  for (int v = tid; v < VOCAB; v += 256)
    orow[v] = (v == amax) ? 1.0f : 0.0f;

  const float* erow = emb_W + (size_t)amax * EMB;
  bf16_t* xrow = x_bf + (size_t)b * EMB;
  for (int e = tid; e < EMB; e += 256)
    xrow[e] = (bf16_t)erow[e];
}

// ---------------------------------------------------------------------------
// Setup kernels
// ---------------------------------------------------------------------------
__global__ __launch_bounds__(256) void cvt_f32_bf16_kernel(
    const float* __restrict__ src, bf16_t* __restrict__ dst, int n) {
  const int i = blockIdx.x * 256 + threadIdx.x;
  if (i < n) dst[i] = (bf16_t)src[i];
}

__global__ __launch_bounds__(256) void init_state_kernel(
    bf16_t* __restrict__ h_bf, float* __restrict__ c, int n) {
  const int i = blockIdx.x * 256 + threadIdx.x;
  if (i < n) { h_bf[i] = (bf16_t)0.0f; c[i] = 0.0f; }
}

// ---------------------------------------------------------------------------
// Host launcher (deterministic; graph-capture safe: only kernel launches)
// ---------------------------------------------------------------------------
extern "C" void kernel_launch(void* const* d_in, const int* in_sizes, int n_in,
                              void* d_out, int out_size, void* d_ws, size_t ws_size,
                              hipStream_t stream) {
  (void)in_sizes; (void)n_in; (void)out_size; (void)ws_size;

  const float* inputs = (const float*)d_in[0];
  const float* emb_W  = (const float*)d_in[1];
  const float* W_ih   = (const float*)d_in[2];
  const float* W_hh   = (const float*)d_in[3];
  const float* b_ih   = (const float*)d_in[4];
  const float* b_hh   = (const float*)d_in[5];
  const float* W_out  = (const float*)d_in[6];
  const float* b_out  = (const float*)d_in[7];
  const float* gumbel = (const float*)d_in[8];
  float* out = (float*)d_out;

  char* ws = (char*)d_ws;
  size_t off = 0;
  auto alloc = [&](size_t bytes) -> void* {
    void* p = ws + off;
    off = (off + bytes + 255) & ~(size_t)255;
    return p;
  };
  bf16_t* Wih_bf  = (bf16_t*)alloc((size_t)GATES_N * EMB * sizeof(bf16_t)); // 4 MB
  bf16_t* Whh_bf  = (bf16_t*)alloc((size_t)GATES_N * HID * sizeof(bf16_t)); // 8 MB
  bf16_t* Wout_bf = (bf16_t*)alloc((size_t)VOCAB * HID * sizeof(bf16_t));   // 4 MB
  bf16_t* x_bf    = (bf16_t*)alloc((size_t)BATCH * EMB * sizeof(bf16_t));
  bf16_t* h_bf    = (bf16_t*)alloc((size_t)BATCH * HID * sizeof(bf16_t));
  float*  c_buf   = (float*) alloc((size_t)BATCH * HID * sizeof(float));
  float*  gates   = (float*) alloc((size_t)BATCH * GATES_N * sizeof(float)); // 4 MB
  float*  logits  = (float*) alloc((size_t)BATCH * VOCAB * sizeof(float));   // 2 MB

  const int nWih = GATES_N * EMB, nWhh = GATES_N * HID, nWout = VOCAB * HID;
  const int nX = BATCH * EMB, nState = BATCH * HID;

  // Per-call weight conversion + state init — bf16 weights stay hot in L2.
  cvt_f32_bf16_kernel<<<(nWih  + 255) / 256, 256, 0, stream>>>(W_ih,  Wih_bf,  nWih);
  cvt_f32_bf16_kernel<<<(nWhh  + 255) / 256, 256, 0, stream>>>(W_hh,  Whh_bf,  nWhh);
  cvt_f32_bf16_kernel<<<(nWout + 255) / 256, 256, 0, stream>>>(W_out, Wout_bf, nWout);
  cvt_f32_bf16_kernel<<<(nX    + 255) / 256, 256, 0, stream>>>(inputs, x_bf,   nX);
  init_state_kernel  <<<(nState+ 255) / 256, 256, 0, stream>>>(h_bf, c_buf, nState);

  for (int s = 0; s < SEQ_LEN; ++s) {
    gates_gemm_kernel<<<64, 256, 0, stream>>>(x_bf, h_bf, Wih_bf, Whh_bf,
                                              b_ih, b_hh, gates);
    lstm_cell_kernel<<<(BATCH * HID) / 256, 256, 0, stream>>>(gates, c_buf, h_bf);
    logits_gemm_kernel<<<25, 256, 0, stream>>>(h_bf, Wout_bf, b_out,
                                               gumbel + (size_t)s * BATCH * VOCAB,
                                               logits);
    argmax_onehot_kernel<<<BATCH, 256, 0, stream>>>(logits, emb_W, out, x_bf, s);
  }
}
